// multi_SEFraud_Model_64793876627839
// MI455X (gfx1250) — matrix operand order repaired and checked by reading, not compiled, not run on hardware
//
#include <hip/hip_runtime.h>
#include <hip/hip_bf16.h>

// ---------------- problem constants (from reference) ----------------
#define NN 50000
#define FF 128
#define HH 64
#define RR 3
#define TT 2
#define EE 500000
#define CC 2
#define ECHUNK 62500   // 500000 / 8 exactly

typedef __attribute__((ext_vector_type(16))) __bf16 v16bf;
typedef __attribute__((ext_vector_type(8)))  __bf16 v8bf;
typedef __attribute__((ext_vector_type(8)))  float  v8f;

__device__ __forceinline__ __bf16 f2bf(float f) {
  unsigned u = __builtin_bit_cast(unsigned, f);
  unsigned r = u + 0x7FFFu + ((u >> 16) & 1u);
  unsigned short h = (unsigned short)(r >> 16);
  return __builtin_bit_cast(__bf16, h);
}

// ---------------- elementwise / conversion kernels ----------------
__global__ void k_f32_to_bf16(const float* __restrict__ in, __bf16* __restrict__ out, long n) {
  long i = (long)blockIdx.x * blockDim.x + threadIdx.x;
  if (i < n) out[i] = f2bf(in[i]);
}

// W [K,Nc] f32 row-major -> Wt [Nc,K] bf16 row-major
__global__ void k_transpose_w(const float* __restrict__ W, __bf16* __restrict__ Wt, int K, int Nc) {
  int i = blockIdx.x * blockDim.x + threadIdx.x;
  if (i >= K * Nc) return;
  int k = i / Nc, n = i % Nc;
  Wt[(long)n * K + k] = f2bf(W[(long)k * Nc + n]);
}

// ---------------- scatter (segment-sum) kernels ----------------
__global__ void k_scatter_add(const float* __restrict__ feat, const int* __restrict__ src,
                              const int* __restrict__ dst, float* __restrict__ agg,
                              float* __restrict__ cnt, int Ecnt, int Fdim) {
  int e = blockIdx.x * blockDim.x + threadIdx.x;
  if (e >= Ecnt) return;
  int s = src[e], d = dst[e];
  atomicAdd(&cnt[d], 1.0f);
  const float4* fp = (const float4*)(feat + (long)s * Fdim);
  float* ap = agg + (long)d * Fdim;
  for (int f = 0; f < Fdim / 4; ++f) {
    float4 v = fp[f];
    atomicAdd(&ap[4 * f + 0], v.x);
    atomicAdd(&ap[4 * f + 1], v.y);
    atomicAdd(&ap[4 * f + 2], v.z);
    atomicAdd(&ap[4 * f + 3], v.w);
  }
}

__global__ void k_mean_bf(const float* __restrict__ agg, const float* __restrict__ cnt,
                          __bf16* __restrict__ out, int n, int Fdim) {
  long i = (long)blockIdx.x * blockDim.x + threadIdx.x;
  if (i >= (long)n * Fdim) return;
  int node = (int)(i / Fdim);
  out[i] = f2bf(agg[i] / fmaxf(cnt[node], 1.0f));
}

// ---------------- generic bf16 WMMA GEMM ----------------
// out = act(A@Wt^T + bias [+ out]);  A: [M,K] bf16 row-major, Wt: [Nc,K] bf16 row-major.
// Each wave computes a 16-row x 64-col strip: 4 WMMA accumulators sharing one A fragment
// per K-step (4x less A traffic than one-tile-per-wave). grid = (ceil(M/64), Nc/64), block 128.
#define GEMM_ACC  1
#define GEMM_RELU 2
#define GEMM_BIAS 4

__global__ __launch_bounds__(128)
void k_gemm_bf16(const __bf16* __restrict__ A, const __bf16* __restrict__ Wt,
                 const float* __restrict__ bias, float* __restrict__ out,
                 int M, int K, int Nc, int ldc, int col_off, int flags) {
  const int lane = threadIdx.x & 31;
  const int wave = threadIdx.x >> 5;
  const int row0 = (blockIdx.x * 4 + wave) * 16;
  if (row0 >= M) return;                 // uniform per wave -> EXEC stays all-ones for WMMA
  const int colb = blockIdx.y * 64;      // 4 consecutive 16-col tiles
  const int l16 = lane & 15, hi = lane >> 4;
  int arow = row0 + l16; if (arow >= M) arow = M - 1;   // clamp; guarded on store
  const __bf16* Ap = A + (long)arow * K;
  const __bf16* Bp0 = Wt + (long)(colb + 0  + l16) * K + hi * 16;
  const __bf16* Bp1 = Wt + (long)(colb + 16 + l16) * K + hi * 16;
  const __bf16* Bp2 = Wt + (long)(colb + 32 + l16) * K + hi * 16;
  const __bf16* Bp3 = Wt + (long)(colb + 48 + l16) * K + hi * 16;

  v8f c0 = {}, c1 = {}, c2 = {}, c3 = {};
  if (flags & GEMM_ACC) {
#pragma unroll
    for (int v = 0; v < 8; ++v) {
      int r = row0 + v + hi * 8;
      if (r < M) {
        const float* op = out + (long)r * ldc + col_off + colb + l16;
        c0[v] = op[0]; c1[v] = op[16]; c2[v] = op[32]; c3[v] = op[48];
      }
    }
  }
  for (int kb = 0; kb < K; kb += 32) {
    v16bf a = *(const v16bf*)(Ap + kb + hi * 16);      // K = kb+hi*16 .. +15 -> frag elems 0..15
    v16bf b0 = *(const v16bf*)(Bp0 + kb);              // K = kb+hi*16 .. +15 per column
    v16bf b1 = *(const v16bf*)(Bp1 + kb);
    v16bf b2 = *(const v16bf*)(Bp2 + kb);
    v16bf b3 = *(const v16bf*)(Bp3 + kb);
    c0 = __builtin_amdgcn_wmma_f32_16x16x32_bf16(false, a, false, b0, (short)0, c0, false, false);
    c1 = __builtin_amdgcn_wmma_f32_16x16x32_bf16(false, a, false, b1, (short)0, c1, false, false);
    c2 = __builtin_amdgcn_wmma_f32_16x16x32_bf16(false, a, false, b2, (short)0, c2, false, false);
    c3 = __builtin_amdgcn_wmma_f32_16x16x32_bf16(false, a, false, b3, (short)0, c3, false, false);
  }
  float bv0 = 0.f, bv1 = 0.f, bv2 = 0.f, bv3 = 0.f;
  if (flags & GEMM_BIAS) {
    bv0 = bias[colb + 0  + l16];
    bv1 = bias[colb + 16 + l16];
    bv2 = bias[colb + 32 + l16];
    bv3 = bias[colb + 48 + l16];
  }
#pragma unroll
  for (int v = 0; v < 8; ++v) {
    int r = row0 + v + hi * 8;
    if (r < M) {
      float* op = out + (long)r * ldc + col_off + colb + l16;
      float v0 = c0[v] + bv0, v1 = c1[v] + bv1, v2 = c2[v] + bv2, v3 = c3[v] + bv3;
      if (flags & GEMM_RELU) {
        v0 = fmaxf(v0, 0.f); v1 = fmaxf(v1, 0.f); v2 = fmaxf(v2, 0.f); v3 = fmaxf(v3, 0.f);
      }
      op[0] = v0; op[16] = v1; op[32] = v2; op[48] = v3;
    }
  }
}

// ---------------- gating: alpha = softmax_t(trees_t . gW_t + gb_t); lo = y + sum alpha_t*trees_t ----
__global__ void k_gate_combine(const float* __restrict__ t0, const float* __restrict__ t1,
                               const float* __restrict__ y, const float* __restrict__ gW,
                               const float* __restrict__ gb, float* __restrict__ lo,
                               __bf16* __restrict__ lobf, int n) {
  int i = blockIdx.x * blockDim.x + threadIdx.x;
  if (i >= n) return;
  const float* p0 = t0 + (long)i * HH;
  const float* p1 = t1 + (long)i * HH;
  float g0 = gb[0], g1 = gb[1];
  for (int h = 0; h < HH; ++h) { g0 += p0[h] * gW[h]; g1 += p1[h] * gW[HH + h]; }
  float m = fmaxf(g0, g1);
  float e0 = __expf(g0 - m), e1 = __expf(g1 - m);
  float inv = 1.0f / (e0 + e1);
  float a0 = e0 * inv, a1 = e1 * inv;
  const float* py = y + (long)i * HH;
  float* pl = lo + (long)i * HH;
  __bf16* pb = lobf + (long)i * HH;
  for (int h = 0; h < HH; ++h) {
    float v = py[h] + a0 * p0[h] + a1 * p1[h];
    pl[h] = v; pb[h] = f2bf(v);
  }
}

// ---------------- build [lo | x | node_emb] bf16 rows (N x 256) ----------------
__global__ void k_build_hetro(const float* __restrict__ lo, const float* __restrict__ x,
                              const float* __restrict__ nemb, __bf16* __restrict__ out, int n) {
  long i = (long)blockIdx.x * blockDim.x + threadIdx.x;
  if (i >= (long)n * 256) return;
  int node = (int)(i >> 8), j = (int)(i & 255);
  float v;
  if (j < HH)            v = lo[(long)node * HH + j];
  else if (j < HH + FF)  v = x[(long)node * FF + (j - HH)];
  else                   v = nemb[j - HH - FF];
  out[i] = f2bf(v);
}

__global__ void k_mask_mul(const float* __restrict__ x, const float* __restrict__ fm,
                           __bf16* __restrict__ mxbf, long n) {
  long i = (long)blockIdx.x * blockDim.x + threadIdx.x;
  if (i < n) mxbf[i] = f2bf(x[i] * fm[i]);
}

// ---------------- build [lo[src] | lo[dst] | edge_emb] bf16 rows for an edge chunk ----------------
__global__ void k_build_ef(const float* __restrict__ lo, const float* __restrict__ eemb,
                           const int* __restrict__ src, const int* __restrict__ dst,
                           int e0, int cnt, __bf16* __restrict__ out) {
  long i = (long)blockIdx.x * blockDim.x + threadIdx.x;
  if (i >= (long)cnt * 192) return;
  int el = (int)(i / 192), j = (int)(i % 192);
  int e = e0 + el;
  float v;
  if (j < HH)            v = lo[(long)src[e] * HH + j];
  else if (j < 2 * HH)   v = lo[(long)dst[e] * HH + (j - HH)];
  else                   v = eemb[j - 2 * HH];
  out[(long)el * 192 + j] = f2bf(v);
}

// em = h @ W2 + b2  (h is already relu'd by GEMM epilogue)
__global__ void k_edge_dot(const float* __restrict__ h, const float* __restrict__ W2,
                           const float* __restrict__ b2, float* __restrict__ em, int e0, int cnt) {
  int i = blockIdx.x * blockDim.x + threadIdx.x;
  if (i >= cnt) return;
  const float4* hp = (const float4*)(h + (long)i * FF);
  const float4* wp = (const float4*)W2;
  float s = b2[0];
  for (int k = 0; k < FF / 4; ++k) {
    float4 hv = hp[k], wv = wp[k];
    s += hv.x * wv.x + hv.y * wv.y + hv.z * wv.z + hv.w * wv.w;
  }
  em[e0 + i] = s;
}

// ---------------- min/max of em via order-preserving uint encoding ----------------
__device__ __forceinline__ unsigned fenc(float f) {
  unsigned u = __builtin_bit_cast(unsigned, f);
  return (u & 0x80000000u) ? ~u : (u | 0x80000000u);
}
__device__ __forceinline__ float fdec(unsigned k) {
  unsigned u = (k & 0x80000000u) ? (k & 0x7FFFFFFFu) : ~k;
  return __builtin_bit_cast(float, u);
}
__global__ void k_init_mm(unsigned* mm) { mm[0] = 0xFFFFFFFFu; mm[1] = 0u; }
__global__ void k_minmax(const float* __restrict__ em, int n, unsigned* __restrict__ mm) {
  int i = blockIdx.x * blockDim.x + threadIdx.x;
  if (i >= n) return;
  unsigned k = fenc(em[i]);
  atomicMin(&mm[0], k);
  atomicMax(&mm[1], k);
}
__global__ void k_edge_w(const float* __restrict__ em, const unsigned* __restrict__ mm,
                         float* __restrict__ w, int n) {
  int i = blockIdx.x * blockDim.x + threadIdx.x;
  if (i >= n) return;
  float mn = fdec(mm[0]), mx = fdec(mm[1]);
  float den = mx - mn;
  den = (fabsf(den) < 1e-20f) ? 1e-20f : den;
  w[i] = (em[i] - mn) / den;
}

// ---------------- GCN pieces ----------------
__global__ void k_scatter_deg(const float* __restrict__ w, const int* __restrict__ dst,
                              float* __restrict__ deg, int Ecnt) {
  int e = blockIdx.x * blockDim.x + threadIdx.x;
  if (e < Ecnt) atomicAdd(&deg[dst[e]], w[e]);
}
__global__ void k_add1(float* deg, int n) {
  int i = blockIdx.x * blockDim.x + threadIdx.x;
  if (i < n) deg[i] += 1.0f;
}
__global__ void k_scatter_gcn(const float* __restrict__ xw, const int* __restrict__ src,
                              const int* __restrict__ dst, const float* __restrict__ w,
                              const float* __restrict__ deg, float* __restrict__ zacc, int Ecnt) {
  int e = blockIdx.x * blockDim.x + threadIdx.x;
  if (e >= Ecnt) return;
  int s = src[e], d = dst[e];
  float cf = w[e] * rsqrtf(deg[s]) * rsqrtf(deg[d]);
  const float4* xp = (const float4*)(xw + (long)s * HH);
  float* zp = zacc + (long)d * HH;
  for (int h = 0; h < HH / 4; ++h) {
    float4 v = xp[h];
    atomicAdd(&zp[4 * h + 0], cf * v.x);
    atomicAdd(&zp[4 * h + 1], cf * v.y);
    atomicAdd(&zp[4 * h + 2], cf * v.z);
    atomicAdd(&zp[4 * h + 3], cf * v.w);
  }
}
__global__ void k_gcn_combine(const float* __restrict__ zacc, const float* __restrict__ xw,
                              const float* __restrict__ deg, const float* __restrict__ bias,
                              float* __restrict__ out, int ldo, int col_off, int relu,
                              __bf16* __restrict__ obf, int n) {
  long i = (long)blockIdx.x * blockDim.x + threadIdx.x;
  if (i >= (long)n * HH) return;
  int node = (int)(i >> 6), h = (int)(i & 63);
  float v = zacc[i] + xw[i] / deg[node] + bias[h];
  if (relu) v = fmaxf(v, 0.0f);
  out[(long)node * ldo + col_off + h] = v;
  if (obf) obf[i] = f2bf(v);
}

// ---------------- head: logits = clip(x_temp @ lin_W + lin_b); log_softmax ----------------
__global__ void k_head(const float* __restrict__ xt, const float* __restrict__ W,
                       const float* __restrict__ b, float* __restrict__ out, int n) {
  int i = blockIdx.x * blockDim.x + threadIdx.x;
  if (i >= n) return;
  const float* xp = xt + (long)i * (RR * HH);
  float l0 = b[0], l1 = b[1];
  for (int j = 0; j < RR * HH; ++j) {
    float v = xp[j];
    l0 += v * W[j * 2 + 0];
    l1 += v * W[j * 2 + 1];
  }
  l0 = fminf(fmaxf(l0, -1e10f), 1e10f);
  l1 = fminf(fmaxf(l1, -1e10f), 1e10f);
  float m = fmaxf(l0, l1);
  float lse = m + __logf(__expf(l0 - m) + __expf(l1 - m));
  out[(long)i * 2 + 0] = l0 - lse;
  out[(long)i * 2 + 1] = l1 - lse;
}

// ---------------- host driver ----------------
extern "C" void kernel_launch(void* const* d_in, const int* in_sizes, int n_in,
                              void* d_out, int out_size, void* d_ws, size_t ws_size,
                              hipStream_t stream) {
  const float* x         = (const float*)d_in[0];
  const int*   edge_main = (const int*)  d_in[1];
  const int*   edge_tree = (const int*)  d_in[2];
  const float* sage_Wl0  = (const float*)d_in[3];
  const float* sage_Wr0  = (const float*)d_in[4];
  const float* sage_b0   = (const float*)d_in[5];
  const float* sage_Wl1  = (const float*)d_in[6];
  const float* sage_Wr1  = (const float*)d_in[7];
  const float* sage_b1   = (const float*)d_in[8];
  const float* tree_Wl   = (const float*)d_in[9];
  const float* tree_Wr   = (const float*)d_in[10];
  const float* tree_b    = (const float*)d_in[11];
  const float* gate_W    = (const float*)d_in[12];
  const float* gate_b    = (const float*)d_in[13];
  const float* node_emb  = (const float*)d_in[14];
  const float* edge_emb  = (const float*)d_in[15];
  const float* fm_W      = (const float*)d_in[16];
  const float* fm_b      = (const float*)d_in[17];
  const float* ea_W1     = (const float*)d_in[18];
  const float* ea_b1     = (const float*)d_in[19];
  const float* ea_W2     = (const float*)d_in[20];
  const float* ea_b2     = (const float*)d_in[21];
  const float* gcn_W0    = (const float*)d_in[22];
  const float* gcn_b0    = (const float*)d_in[23];
  const float* gcn_W1    = (const float*)d_in[24];
  const float* gcn_b1    = (const float*)d_in[25];
  const float* lin_W     = (const float*)d_in[26];
  const float* lin_b     = (const float*)d_in[27];

  float* out_logp = (float*)d_out;
  float* x_temp   = out_logp + (size_t)NN * CC;   // [N, R*H] second output region

  // ---- workspace carve-up (256B aligned) ----
  char* base = (char*)d_ws;
  size_t off = 0;
  auto alloc = [&](size_t bytes) -> void* {
    void* p = base + off;
    off = (off + bytes + 255) & ~(size_t)255;
    return p;
  };
  __bf16* x_bf     = (__bf16*)alloc((size_t)NN * FF * 2);
  float*  agg      = (float*) alloc((size_t)NN * FF * 4);   // also fm output [N,128]
  float*  cnt      = (float*) alloc((size_t)NN * 4);
  __bf16* mean_bf  = (__bf16*)alloc((size_t)NN * FF * 2);
  float*  y1       = (float*) alloc((size_t)NN * HH * 4);
  __bf16* y1_bf    = (__bf16*)alloc((size_t)NN * HH * 2);
  float*  ybuf     = (float*) alloc((size_t)NN * HH * 4);
  float*  t0buf    = (float*) alloc((size_t)NN * HH * 4);
  float*  t1buf    = (float*) alloc((size_t)NN * HH * 4);
  float*  lo       = (float*) alloc((size_t)NN * HH * 4);
  __bf16* lo_bf    = (__bf16*)alloc((size_t)NN * HH * 2);
  __bf16* hetro_bf = (__bf16*)alloc((size_t)NN * 256 * 2);
  __bf16* mx_bf    = (__bf16*)alloc((size_t)NN * FF * 2);
  __bf16* ef_bf    = (__bf16*)alloc((size_t)ECHUNK * 192 * 2);
  float*  h_chunk  = (float*) alloc((size_t)ECHUNK * FF * 4);
  float*  em       = (float*) alloc((size_t)EE * 4);
  float*  w_e      = (float*) alloc((size_t)EE * 4);
  unsigned* mm     = (unsigned*)alloc(256);
  float*  deg      = (float*) alloc((size_t)NN * 4);
  float*  xw       = (float*) alloc((size_t)NN * HH * 4);
  float*  zacc     = (float*) alloc((size_t)NN * HH * 4);
  float*  z0       = (float*) alloc((size_t)NN * HH * 4);
  __bf16* z0_bf    = (__bf16*)alloc((size_t)NN * HH * 2);
  __bf16* wt_node  = (__bf16*)alloc((size_t)256 * 128 * 2);  // largest transposed weight
  __bf16* wt_ea    = (__bf16*)alloc((size_t)128 * 192 * 2);
  (void)ws_size; (void)in_sizes; (void)n_in; (void)out_size;

  auto cdiv = [](long a, long b) { return (int)((a + b - 1) / b); };
  auto gemm = [&](const __bf16* A, const __bf16* Wt, const float* bias, float* out,
                  int M, int K, int Nc, int ldc, int coff, int flags) {
    dim3 g(cdiv(M, 64), Nc / 64);   // each wave: 16 rows x 64 cols (4 WMMA tiles)
    k_gemm_bf16<<<g, 128, 0, stream>>>(A, Wt, bias, out, M, K, Nc, ldc, coff, flags);
  };
  auto transpose = [&](const float* W, __bf16* Wt, int K, int Nc) {
    k_transpose_w<<<cdiv((long)K * Nc, 256), 256, 0, stream>>>(W, Wt, K, Nc);
  };

  k_f32_to_bf16<<<cdiv((long)NN * FF, 256), 256, 0, stream>>>(x, x_bf, (long)NN * FF);
  transpose(ea_W1, wt_ea, 192, 128);   // shared across relations

  for (int r = 0; r < RR; ++r) {
    const int* src = edge_main + (size_t)r * 2 * EE;
    const int* dst = src + EE;

    // ---- SAGE layer 0: y1 = relu(mean@Wl0 + x@Wr0 + b0) ----
    hipMemsetAsync(agg, 0, (size_t)NN * FF * 4, stream);
    hipMemsetAsync(cnt, 0, (size_t)NN * 4, stream);
    k_scatter_add<<<cdiv(EE, 256), 256, 0, stream>>>(x, src, dst, agg, cnt, EE, FF);
    k_mean_bf<<<cdiv((long)NN * FF, 256), 256, 0, stream>>>(agg, cnt, mean_bf, NN, FF);
    transpose(sage_Wl0 + (size_t)r * FF * HH, wt_node, FF, HH);
    gemm(mean_bf, wt_node, sage_b0 + (size_t)r * HH, y1, NN, FF, HH, HH, 0, GEMM_BIAS);
    transpose(sage_Wr0 + (size_t)r * FF * HH, wt_node, FF, HH);
    gemm(x_bf, wt_node, nullptr, y1, NN, FF, HH, HH, 0, GEMM_ACC | GEMM_RELU);
    k_f32_to_bf16<<<cdiv((long)NN * HH, 256), 256, 0, stream>>>(y1, y1_bf, (long)NN * HH);

    // ---- SAGE layer 1: y = mean(y1)@Wl1 + y1@Wr1 + b1 ----
    hipMemsetAsync(agg, 0, (size_t)NN * HH * 4, stream);
    hipMemsetAsync(cnt, 0, (size_t)NN * 4, stream);
    k_scatter_add<<<cdiv(EE, 256), 256, 0, stream>>>(y1, src, dst, agg, cnt, EE, HH);
    k_mean_bf<<<cdiv((long)NN * HH, 256), 256, 0, stream>>>(agg, cnt, mean_bf, NN, HH);
    transpose(sage_Wl1 + (size_t)r * HH * HH, wt_node, HH, HH);
    gemm(mean_bf, wt_node, sage_b1 + (size_t)r * HH, ybuf, NN, HH, HH, HH, 0, GEMM_BIAS);
    transpose(sage_Wr1 + (size_t)r * HH * HH, wt_node, HH, HH);
    gemm(y1_bf, wt_node, nullptr, ybuf, NN, HH, HH, HH, 0, GEMM_ACC);

    // ---- tree branches ----
    float* tbufs[TT] = { t0buf, t1buf };
    for (int t = 0; t < TT; ++t) {
      const int* ts = edge_tree + ((size_t)(r * TT + t) * 2) * EE;
      const int* td = ts + EE;
      hipMemsetAsync(agg, 0, (size_t)NN * FF * 4, stream);
      hipMemsetAsync(cnt, 0, (size_t)NN * 4, stream);
      k_scatter_add<<<cdiv(EE, 256), 256, 0, stream>>>(x, ts, td, agg, cnt, EE, FF);
      k_mean_bf<<<cdiv((long)NN * FF, 256), 256, 0, stream>>>(agg, cnt, mean_bf, NN, FF);
      transpose(tree_Wl + (size_t)(r * TT + t) * FF * HH, wt_node, FF, HH);
      gemm(mean_bf, wt_node, tree_b + (size_t)(r * TT + t) * HH, tbufs[t], NN, FF, HH, HH, 0, GEMM_BIAS);
      transpose(tree_Wr + (size_t)(r * TT + t) * FF * HH, wt_node, FF, HH);
      gemm(x_bf, wt_node, nullptr, tbufs[t], NN, FF, HH, HH, 0, GEMM_ACC | GEMM_RELU);
    }
    k_gate_combine<<<cdiv(NN, 128), 128, 0, stream>>>(t0buf, t1buf, ybuf,
        gate_W + (size_t)r * TT * HH, gate_b + (size_t)r * TT, lo, lo_bf, NN);

    // ---- feature mask: fm = [lo|x|node_emb]@fm_W + fm_b ; mx = x*fm ----
    k_build_hetro<<<cdiv((long)NN * 256, 256), 256, 0, stream>>>(lo, x, node_emb + (size_t)r * HH, hetro_bf, NN);
    transpose(fm_W + (size_t)r * 256 * FF, wt_node, 256, FF);
    gemm(hetro_bf, wt_node, fm_b + (size_t)r * FF, agg /*fm*/, NN, 256, FF, FF, 0, GEMM_BIAS);
    k_mask_mul<<<cdiv((long)NN * FF, 256), 256, 0, stream>>>(x, agg, mx_bf, (long)NN * FF);

    // ---- edge MLP in chunks: em = relu([lo_s|lo_d|eemb]@W1+b1)@W2+b2 ----
    for (int cch = 0; cch < EE / ECHUNK; ++cch) {
      int e0 = cch * ECHUNK;
      k_build_ef<<<cdiv((long)ECHUNK * 192, 256), 256, 0, stream>>>(lo, edge_emb + (size_t)r * HH,
                                                                    src, dst, e0, ECHUNK, ef_bf);
      gemm(ef_bf, wt_ea, ea_b1, h_chunk, ECHUNK, 192, FF, FF, 0, GEMM_BIAS | GEMM_RELU);
      k_edge_dot<<<cdiv(ECHUNK, 256), 256, 0, stream>>>(h_chunk, ea_W2, ea_b2, em, e0, ECHUNK);
    }
    k_init_mm<<<1, 1, 0, stream>>>(mm);
    k_minmax<<<cdiv(EE, 256), 256, 0, stream>>>(em, EE, mm);
    k_edge_w<<<cdiv(EE, 256), 256, 0, stream>>>(em, mm, w_e, EE);

    // ---- GCN layer 0 (input mx, relu) ----
    hipMemsetAsync(deg, 0, (size_t)NN * 4, stream);
    k_scatter_deg<<<cdiv(EE, 256), 256, 0, stream>>>(w_e, dst, deg, EE);
    k_add1<<<cdiv(NN, 256), 256, 0, stream>>>(deg, NN);
    transpose(gcn_W0 + (size_t)r * FF * HH, wt_node, FF, HH);
    gemm(mx_bf, wt_node, nullptr, xw, NN, FF, HH, HH, 0, 0);
    hipMemsetAsync(zacc, 0, (size_t)NN * HH * 4, stream);
    k_scatter_gcn<<<cdiv(EE, 256), 256, 0, stream>>>(xw, src, dst, w_e, deg, zacc, EE);
    k_gcn_combine<<<cdiv((long)NN * HH, 256), 256, 0, stream>>>(zacc, xw, deg,
        gcn_b0 + (size_t)r * HH, z0, HH, 0, 1, z0_bf, NN);

    // ---- GCN layer 1 (no relu) -> x_temp columns [r*H, (r+1)*H) ----
    transpose(gcn_W1 + (size_t)r * HH * HH, wt_node, HH, HH);
    gemm(z0_bf, wt_node, nullptr, xw, NN, HH, HH, HH, 0, 0);
    hipMemsetAsync(zacc, 0, (size_t)NN * HH * 4, stream);
    k_scatter_gcn<<<cdiv(EE, 256), 256, 0, stream>>>(xw, src, dst, w_e, deg, zacc, EE);
    k_gcn_combine<<<cdiv((long)NN * HH, 256), 256, 0, stream>>>(zacc, xw, deg,
        gcn_b1 + (size_t)r * HH, x_temp, RR * HH, r * HH, 0, nullptr, NN);
  }

  // ---- head: log_softmax(clip(x_temp @ lin_W + lin_b)) ----
  k_head<<<cdiv(NN, 128), 128, 0, stream>>>(x_temp, lin_W, lin_b, out_logp, NN);
}
